// SimpleGIN_24721831756436
// MI455X (gfx1250) — matrix operand order repaired and checked by reading, not compiled, not run on hardware
//
#include <hip/hip_runtime.h>

#define HID 128

typedef __attribute__((ext_vector_type(16))) _Float16 v16h;
typedef __attribute__((ext_vector_type(8)))  _Float16 v8h;
typedef __attribute__((ext_vector_type(8)))  float    v8f;

// h[n][:] = emb[clamp(x[n])][:]
__global__ __launch_bounds__(256) void gin_embed(
    const int* __restrict__ x, const float* __restrict__ emb,
    float* __restrict__ h, int n_nodes, int emb_rows)
{
  int t = blockIdx.x * blockDim.x + threadIdx.x;
  int n = t >> 5, c4 = t & 31;
  if (n >= n_nodes) return;
  int r = x[n];
  r = r < 0 ? 0 : (r >= emb_rows ? emb_rows - 1 : r);
  ((float4*)(h + (size_t)n * HID))[c4] = ((const float4*)(emb + (size_t)r * HID))[c4];
}

__global__ __launch_bounds__(256) void copy_f4(
    const float* __restrict__ src, float* __restrict__ dst, int n4)
{
  int t = blockIdx.x * blockDim.x + threadIdx.x;
  if (t < n4) ((float4*)dst)[t] = ((const float4*)src)[t];
}

__global__ __launch_bounds__(256) void zero_f(float* __restrict__ p, int n)
{
  int t = blockIdx.x * blockDim.x + threadIdx.x;
  if (t < n) p[t] = 0.0f;
}

// z[dst[e]] += h[src[e]] : one wave per edge, scalar edge indices,
// lane = 4-channel group -> one coalesced b128 load + 4 L2 atomics per lane.
__global__ __launch_bounds__(256) void edge_scatter(
    const int* __restrict__ src, const int* __restrict__ dst,
    const float* __restrict__ h, float* __restrict__ z, int n_edges)
{
  int e = __builtin_amdgcn_readfirstlane(blockIdx.x * 8 + (threadIdx.x >> 5));
  int lane = threadIdx.x & 31;
  if (e >= n_edges) return;
  int s = __builtin_amdgcn_readfirstlane(src[e]);
  int d = __builtin_amdgcn_readfirstlane(dst[e]);
  float4 v = ((const float4*)(h + (size_t)s * HID))[lane];
  float* zp = z + (size_t)d * HID + (lane << 2);
  atomicAdd(zp + 0, v.x);
  atomicAdd(zp + 1, v.y);
  atomicAdd(zp + 2, v.z);
  atomicAdd(zp + 3, v.w);
}

// Pack W[128][128] (row-major, out = in @ W) into WMMA B-fragment order:
// Wp[(((kt*8 + jt)*32) + lane)*16 + i] = (f16) W[kt*32 + 16*(lane/16) + i][jt*16 + lane%16]
__global__ __launch_bounds__(256) void pack_w(
    const float* __restrict__ W, _Float16* __restrict__ Wp)
{
  int idx = blockIdx.x * blockDim.x + threadIdx.x;
  if (idx >= 4 * 8 * 32 * 16) return;
  int i    = idx & 15;
  int lane = (idx >> 4) & 31;
  int jt   = (idx >> 9) & 7;
  int kt   = (idx >> 12) & 3;
  int j = jt * 16 + (lane & 15);
  int k = kt * 32 + ((lane >> 4) << 4) + i;
  Wp[idx] = (_Float16)W[k * HID + j];
}

// C[nrows x 128] = act(A[nrows x 128] @ W + bias), W pre-packed as B fragments.
// Block = 16 rows x 128 cols; 8 waves, one 16-col tile per wave; K=128 via 4 WMMAs.
// A tile converted to f16 once into LDS in A-fragment lane order; steady-state
// loop = 1 v16h DS load + 1 v16h global load + v_wmma. Straight-line epilogue.
template <bool DO_RELU>
__global__ __launch_bounds__(256) void gemm_wmma(
    const float* __restrict__ A, const _Float16* __restrict__ Wp,
    const float* __restrict__ bias, float* __restrict__ C, int nrows)
{
  // Apk[kt][lane][i] : A 16x32 fragment per k-tile, contiguous k = 16*(lane/16) + i (4 KB)
  __shared__ _Float16 Apk[4 * 32 * 16];

  const int r0  = blockIdx.x * 16;
  const int tid = threadIdx.x;
  {
    // thread -> (kt, lane, half-of-fragment); writes 8 halves = one b128 DS store
    int kt   = tid >> 6;          // 0..3
    int ln   = (tid >> 1) & 31;   // fragment lane 0..31
    int part = tid & 1;           // halves 0..7 or 8..15
    int row  = ln & 15;           // M = lane % 16
    int koff = (ln >> 4) << 4;    // +16 K for upper half-wave (contiguous A layout)
    int col  = kt * 32 + koff + (part << 3);
    int rg = r0 + row; if (rg >= nrows) rg = nrows - 1;
    const float4* s4 = (const float4*)(A + (size_t)rg * HID + col);
    float4 v0 = s4[0], v1 = s4[1];
    v8h hv;
    hv[0] = (_Float16)v0.x; hv[1] = (_Float16)v0.y;
    hv[2] = (_Float16)v0.z; hv[3] = (_Float16)v0.w;
    hv[4] = (_Float16)v1.x; hv[5] = (_Float16)v1.y;
    hv[6] = (_Float16)v1.z; hv[7] = (_Float16)v1.w;
    *(v8h*)&Apk[((kt * 32 + ln) << 4) + (part << 3)] = hv;
  }
  __syncthreads();

  const int wave  = tid >> 5;       // column tile 0..7
  const int lane  = tid & 31;
  const int j0    = wave * 16;
  const int mHalf = lane >> 4;
  const int mIdx  = lane & 15;

  v8f acc = {};
#pragma unroll
  for (int kt = 0; kt < 4; ++kt) {
    v16h a = *(const v16h*)&Apk[(kt * 32 + lane) << 4];
    v16h b = *(const v16h*)(Wp + ((size_t)((kt * 8 + wave) * 32 + lane) << 4));
    acc = __builtin_amdgcn_wmma_f32_16x16x32_f16(
        false, a, false, b, (short)0, acc, false, false);
  }

  // C/D layout: VGPR r -> row r (lanes 0-15) or r+8 (lanes 16-31), col = lane%16
  float bv = bias[j0 + mIdx];
  float* cp = C + (size_t)(r0 + (mHalf << 3)) * HID + j0 + mIdx;
  if (r0 + 16 <= nrows) {
    // full tile: 8 straight-line stores at immediate offsets (r*512B)
#pragma unroll
    for (int r = 0; r < 8; ++r) {
      float v = acc[r] + bv;
      if (DO_RELU) v = fmaxf(v, 0.0f);
      cp[(size_t)r * HID] = v;
    }
  } else {
#pragma unroll
    for (int r = 0; r < 8; ++r) {
      int row = r0 + r + (mHalf << 3);
      if (row < nrows) {
        float v = acc[r] + bv;
        if (DO_RELU) v = fmaxf(v, 0.0f);
        cp[(size_t)r * HID] = v;
      }
    }
  }
}

// pooled[batch[n]] += h[n]; cnt[batch[n]] += 1  (wave per node)
__global__ __launch_bounds__(256) void pool_scatter(
    const int* __restrict__ batch, const float* __restrict__ h,
    float* __restrict__ pooled, float* __restrict__ cnt, int n_nodes)
{
  int n = __builtin_amdgcn_readfirstlane(blockIdx.x * 8 + (threadIdx.x >> 5));
  int lane = threadIdx.x & 31;
  if (n >= n_nodes) return;
  int g = __builtin_amdgcn_readfirstlane(batch[n]);
  float4 v = ((const float4*)(h + (size_t)n * HID))[lane];
  float* pp = pooled + (size_t)g * HID + (lane << 2);
  atomicAdd(pp + 0, v.x);
  atomicAdd(pp + 1, v.y);
  atomicAdd(pp + 2, v.z);
  atomicAdd(pp + 3, v.w);
  if (lane == 0) atomicAdd(cnt + g, 1.0f);
}

// out[g][o] = dot(pooled_sum[g], Wlin[:,o]) / max(cnt,1) + blin[o]
__global__ __launch_bounds__(256) void final_linear(
    const float* __restrict__ pooled, const float* __restrict__ cnt,
    const float* __restrict__ Wlin, const float* __restrict__ blin,
    float* __restrict__ out, int n_graphs, int out_dim)
{
  int t = blockIdx.x * blockDim.x + threadIdx.x;
  if (t >= n_graphs * out_dim) return;
  int g = t / out_dim, o = t % out_dim;
  float c = cnt[g]; c = c > 1.0f ? c : 1.0f;
  const float* p = pooled + (size_t)g * HID;
  float s = 0.0f;
#pragma unroll 4
  for (int k = 0; k < HID; ++k) s += p[k] * Wlin[k * out_dim + o];
  out[t] = s / c + blin[o];
}

extern "C" void kernel_launch(void* const* d_in, const int* in_sizes, int n_in,
                              void* d_out, int out_size, void* d_ws, size_t ws_size,
                              hipStream_t stream)
{
  const int*   x     = (const int*)d_in[0];
  const int*   ei    = (const int*)d_in[1];
  const int*   batch = (const int*)d_in[2];
  const float* emb   = (const float*)d_in[3];
  const float* W1a   = (const float*)d_in[4];
  const float* b1a   = (const float*)d_in[5];
  const float* W1b   = (const float*)d_in[6];
  const float* b1b   = (const float*)d_in[7];
  const float* W2a   = (const float*)d_in[8];
  const float* b2a   = (const float*)d_in[9];
  const float* W2b   = (const float*)d_in[10];
  const float* b2b   = (const float*)d_in[11];
  const float* Wlin  = (const float*)d_in[12];
  const float* blin  = (const float*)d_in[13];

  const int n_nodes  = in_sizes[0];
  const int n_edges  = in_sizes[1] / 2;
  const int emb_rows = in_sizes[3] / HID;
  const int out_dim  = in_sizes[13];
  const int ng       = out_size / out_dim;

  const int* esrc = ei;             // edge_index[0]
  const int* edst = ei + n_edges;   // edge_index[1]

  char*  ws   = (char*)d_ws;
  size_t szNH = (size_t)n_nodes * HID * sizeof(float);
  float*    h      = (float*)(ws);
  float*    z      = (float*)(ws + szNH);
  float*    t      = (float*)(ws + 2 * szNH);
  _Float16* Wp     = (_Float16*)(ws + 3 * szNH);                       // 4 * 16384 halves
  float*    pooled = (float*)(ws + 3 * szNH + 4 * 16384 * sizeof(_Float16));
  float*    cnt    = pooled + (size_t)ng * HID;

  const int B = 256;
  int pk_blk   = (4 * 8 * 32 * 16 + B - 1) / B;
  int n4       = n_nodes * 32;                // float4 elements per node buffer
  int cpy_blk  = (n4 + B - 1) / B;
  int edge_blk = (n_edges + 7) / 8;           // wave per edge, 8 waves per block
  int node_blk = (n_nodes + 7) / 8;           // wave per node
  int gemm_blk = (n_nodes + 15) / 16;

  // Pre-pack weights into WMMA B-fragment layout (f16)
  pack_w<<<pk_blk, B, 0, stream>>>(W1a, Wp + 0 * 16384);
  pack_w<<<pk_blk, B, 0, stream>>>(W1b, Wp + 1 * 16384);
  pack_w<<<pk_blk, B, 0, stream>>>(W2a, Wp + 2 * 16384);
  pack_w<<<pk_blk, B, 0, stream>>>(W2b, Wp + 3 * 16384);

  // h = emb[x]
  gin_embed<<<cpy_blk, B, 0, stream>>>(x, emb, h, n_nodes, emb_rows);

  // ---- GIN layer 1 ----
  copy_f4<<<cpy_blk, B, 0, stream>>>(h, z, n4);                            // z = h
  edge_scatter<<<edge_blk, B, 0, stream>>>(esrc, edst, h, z, n_edges);     // z += agg
  gemm_wmma<true><<<gemm_blk, B, 0, stream>>>(z, Wp + 0 * 16384, b1a, t, n_nodes);
  gemm_wmma<true><<<gemm_blk, B, 0, stream>>>(t, Wp + 1 * 16384, b1b, h, n_nodes);

  // ---- GIN layer 2 ----
  copy_f4<<<cpy_blk, B, 0, stream>>>(h, z, n4);
  edge_scatter<<<edge_blk, B, 0, stream>>>(esrc, edst, h, z, n_edges);
  gemm_wmma<true><<<gemm_blk, B, 0, stream>>>(z, Wp + 2 * 16384, b2a, t, n_nodes);
  gemm_wmma<false><<<gemm_blk, B, 0, stream>>>(t, Wp + 3 * 16384, b2b, h, n_nodes);

  // ---- mean pool + linear head ----
  int zn = ng * HID + ng;
  zero_f<<<(zn + B - 1) / B, B, 0, stream>>>(pooled, zn);
  pool_scatter<<<node_blk, B, 0, stream>>>(batch, h, pooled, cnt, n_nodes);
  final_linear<<<(ng * out_dim + B - 1) / B, B, 0, stream>>>(
      pooled, cnt, Wlin, blin, (float*)d_out, ng, out_dim);
}
